// GAT_14980845929104
// MI455X (gfx1250) — matrix operand order repaired and checked by reading, not compiled, not run on hardware
//
#include <hip/hip_runtime.h>
#include <hip/hip_bf16.h>

typedef __attribute__((ext_vector_type(16))) _Float16 v16h;
typedef __attribute__((ext_vector_type(8)))  _Float16 v8h;
typedef __attribute__((ext_vector_type(8)))  float    v8f;

#define GAT_HEADS 4
#define NEG_SLOPE 0.2f

// ---------- monotone float<->uint encoding for atomic segment-max ----------
__device__ __forceinline__ unsigned f_enc(float f) {
    unsigned u = __float_as_uint(f);
    return (u & 0x80000000u) ? ~u : (u | 0x80000000u);
}
__device__ __forceinline__ float f_dec(unsigned o) {
    unsigned u = (o & 0x80000000u) ? (o & 0x7fffffffu) : ~o;
    return __uint_as_float(u);
}

// ---------- f32 -> f16 convert (activations, row-major) ----------
__global__ void k_f32_to_f16(const float* __restrict__ in, _Float16* __restrict__ out, int n) {
    int i = blockIdx.x * blockDim.x + threadIdx.x;
    if (i < n) out[i] = (_Float16)in[i];
}

// ---------- W[K,M] f32 -> WT[M,K] f16 (transpose so B-fragments are contiguous) ----------
__global__ void k_w_to_f16t(const float* __restrict__ W, _Float16* __restrict__ WT,
                            int K, int M) {
    int i = blockIdx.x * blockDim.x + threadIdx.x;   // i = m*K + k
    if (i >= K * M) return;
    int m = i / K;
    int k = i - m * K;
    WT[i] = (_Float16)W[(size_t)k * M + m];
}

// ---------- WMMA GEMM: C[N,M] = A[N,K] * W[K,M] with W given transposed WT[M,K] ----------
// One wave computes a 16 x (16*CPW) strip: shared A fragment, CPW accumulators.
// All fragment loads are contiguous 16B (global_load_b128). N%16==0, M%(16*CPW)==0, K%32==0.
template <int CPW>
__global__ void k_gemm_wmma(const _Float16* __restrict__ A,
                            const _Float16* __restrict__ WT,
                            float* __restrict__ C,
                            int Nrows, int K, int M) {
    int wavesPerBlock = blockDim.x >> 5;
    int wave = blockIdx.x * wavesPerBlock + (threadIdx.x >> 5);
    int lane = threadIdx.x & 31;
    int colGroups = (M >> 4) / CPW;
    int rowTile = wave / colGroups;
    int colTile0 = (wave % colGroups) * CPW;
    if (rowTile * 16 >= Nrows) return;   // wave-uniform exit: EXEC all-1s for WMMA

    int half = lane >> 4;   // which 16-lane half of the wave
    int l16  = lane & 15;

    // A fragment (contiguous k order, same as B; not the ISA table order): lane holds row (lane&15);
    // element i <-> K = kt + half*16 + i  => one contiguous 16-half run
    const _Float16* arow = A + (size_t)(rowTile * 16 + l16) * K + half * 16;

    // B fragment (contiguous k order, same as A; not the ISA table order): lane holds col (lane&15);
    // element i <-> K = kt + half*16 + i  => one contiguous 16-half run of WT row
    const _Float16* brow[CPW];
#pragma unroll
    for (int j = 0; j < CPW; ++j)
        brow[j] = WT + (size_t)((colTile0 + j) * 16 + l16) * K + half * 16;

    v8f acc[CPW];
#pragma unroll
    for (int j = 0; j < CPW; ++j) acc[j] = (v8f){};

    for (int kt = 0; kt < K; kt += 32) {
        v8h alo = *(const v8h*)(arow + kt);
        v8h ahi = *(const v8h*)(arow + kt + 8);
        v16h a = __builtin_shufflevector(alo, ahi, 0,1,2,3,4,5,6,7,8,9,10,11,12,13,14,15);
#pragma unroll
        for (int j = 0; j < CPW; ++j) {
            v8h blo = *(const v8h*)(brow[j] + kt);
            v8h bhi = *(const v8h*)(brow[j] + kt + 8);
            v16h b = __builtin_shufflevector(blo, bhi, 0,1,2,3,4,5,6,7,8,9,10,11,12,13,14,15);
            acc[j] = __builtin_amdgcn_wmma_f32_16x16x32_f16(
                false, a, false, b, (short)0, acc[j], false, false);
        }
    }

    // C/D layout: VGPR r -> row = r + half*8, col = lane&15
#pragma unroll
    for (int j = 0; j < CPW; ++j) {
        float* crow = C + (size_t)(rowTile * 16 + half * 8) * M + (colTile0 + j) * 16 + l16;
#pragma unroll
        for (int r = 0; r < 8; ++r) crow[(size_t)r * M] = acc[j][r];
    }
}

// ---------- per-(node,head) attention coefficients ----------
__global__ void k_attn(const float* __restrict__ Hm, const float* __restrict__ attS,
                       const float* __restrict__ attD, float* __restrict__ asrc,
                       float* __restrict__ adst, int N, int C) {
    int i = blockIdx.x * blockDim.x + threadIdx.x;   // n*HEADS + h
    if (i >= N * GAT_HEADS) return;
    int h = i & (GAT_HEADS - 1);
    int n = i >> 2;
    const float* hp = Hm + (size_t)n * GAT_HEADS * C + (size_t)h * C;
    float s = 0.f, d = 0.f;
    for (int c = 0; c < C; ++c) {
        float v = hp[c];
        s += v * attS[h * C + c];
        d += v * attD[h * C + c];
    }
    asrc[i] = s;
    adst[i] = d;
}

// ---------- init accumulators (workspace is poisoned, must clear every call) ----------
__global__ void k_init(unsigned* __restrict__ amax, float* __restrict__ denom,
                       float* __restrict__ agg, int nh, int nm) {
    int i = blockIdx.x * blockDim.x + threadIdx.x;
    if (i < nh) { amax[i] = 0u; denom[i] = 0.f; }   // 0 < f_enc(any finite alpha)
    if (i < nm) agg[i] = 0.f;
}

// ---------- pass A: segment max of leaky-relu(alpha) over dst ----------
__global__ void k_edge_max(const int* __restrict__ eidx, int E, int N,
                           const float* __restrict__ asrc, const float* __restrict__ adst,
                           unsigned* __restrict__ amax) {
    int i = blockIdx.x * blockDim.x + threadIdx.x;
    int tot = (E + N) * GAT_HEADS;
    if (i >= tot) return;
    int h = i & (GAT_HEADS - 1);
    int e = i >> 2;
    int s, d;
    if (e < E) { s = eidx[e]; d = eidx[E + e]; } else { s = d = e - E; }  // self-loops
    float alpha = asrc[s * GAT_HEADS + h] + adst[d * GAT_HEADS + h];
    alpha = alpha > 0.f ? alpha : alpha * NEG_SLOPE;
    atomicMax(&amax[d * GAT_HEADS + h], f_enc(alpha));
}

// ---------- pass B: e = exp(alpha - amax); denom += e; agg[dst] += h[src]*e ----------
// (coef = e/denom folded into finalize: sum(h*e)/denom == sum(h*e/denom))
template <int C>
__global__ void k_edge_scatter(const int* __restrict__ eidx, int E, int N,
                               const float* __restrict__ asrc, const float* __restrict__ adst,
                               const unsigned* __restrict__ amax, const float* __restrict__ Hm,
                               float* __restrict__ denom, float* __restrict__ agg) {
    int i = blockIdx.x * blockDim.x + threadIdx.x;
    int tot = (E + N) * GAT_HEADS;
    if (i >= tot) return;
    int h = i & (GAT_HEADS - 1);
    int e = i >> 2;
    int s, d;
    if (e < E) { s = eidx[e]; d = eidx[E + e]; } else { s = d = e - E; }
    float alpha = asrc[s * GAT_HEADS + h] + adst[d * GAT_HEADS + h];
    alpha = alpha > 0.f ? alpha : alpha * NEG_SLOPE;
    float ex = __expf(alpha - f_dec(amax[d * GAT_HEADS + h]));
    atomicAdd(&denom[d * GAT_HEADS + h], ex);
    const float* hs = Hm + (size_t)s * GAT_HEADS * C + (size_t)h * C;   // 16B aligned
    float* ag = agg + (size_t)d * GAT_HEADS * C + (size_t)h * C;
    __builtin_prefetch(hs, 0, 1);   // global_prefetch of source feature row
#pragma unroll
    for (int c = 0; c < C; c += 4) {
        float4 v = *(const float4*)(hs + c);
        atomicAdd(&ag[c + 0], v.x * ex);
        atomicAdd(&ag[c + 1], v.y * ex);
        atomicAdd(&ag[c + 2], v.z * ex);
        atomicAdd(&ag[c + 3], v.w * ex);
    }
}

// ---------- finalize (concat layers): out = relu(agg/denom + bias) ----------
__global__ void k_finalize_concat(const float* __restrict__ agg, const float* __restrict__ denom,
                                  const float* __restrict__ bias, float* __restrict__ out,
                                  float* __restrict__ embOut, int N, int C) {
    int HC = GAT_HEADS * C;
    int i = blockIdx.x * blockDim.x + threadIdx.x;   // n*HC + hc
    if (i >= N * HC) return;
    int n = i / HC;
    int hc = i - n * HC;
    int h = hc / C;
    float v = agg[i] / denom[n * GAT_HEADS + h] + bias[hc];
    v = v > 0.f ? v : 0.f;
    out[i] = v;
    if (embOut) embOut[i] = v;
}

// ---------- finalize (final layer): head-mean + bias, then log_softmax ----------
__global__ void k_finalize_mean_lsm(const float* __restrict__ agg, const float* __restrict__ denom,
                                    const float* __restrict__ bias, float* __restrict__ out,
                                    int N, int C) {
    int n = blockIdx.x * blockDim.x + threadIdx.x;
    if (n >= N) return;
    float logit[40];   // C == NCLASS == 40
    float m = -1e30f;
    for (int c = 0; c < C; ++c) {
        float acc = 0.f;
#pragma unroll
        for (int h = 0; h < GAT_HEADS; ++h)
            acc += agg[(size_t)n * GAT_HEADS * C + h * C + c] / denom[n * GAT_HEADS + h];
        float v = acc * (1.0f / GAT_HEADS) + bias[c];
        logit[c] = v;
        m = fmaxf(m, v);
    }
    float ssum = 0.f;
    for (int c = 0; c < C; ++c) ssum += __expf(logit[c] - m);
    float lse = m + __logf(ssum);
    for (int c = 0; c < C; ++c) out[(size_t)n * C + c] = logit[c] - lse;
}

// ---------- one GAT layer ----------
static void launch_layer(const float* Xin, const float* W, const float* attS, const float* attD,
                         const float* bias, const int* eidx, int N, int E, int K, int C,
                         bool isFinal, _Float16* X16, _Float16* WT16, float* Hbuf, float* AGG,
                         float* ASRC, float* ADST, unsigned* AMAX, float* DEN,
                         float* Fout, float* embOut, float* logitsOut, hipStream_t stream) {
    const int M = GAT_HEADS * C;
    const int T = 256;
    int nx = N * K;
    k_f32_to_f16<<<(nx + T - 1) / T, T, 0, stream>>>(Xin, X16, nx);
    int nw = K * M;
    k_w_to_f16t<<<(nw + T - 1) / T, T, 0, stream>>>(W, WT16, K, M);

    const int CPW = 2;   // col tiles per wave; M/32 integral for M=128,160
    int totalWaves = (N / 16) * ((M / 16) / CPW);
    int wavesPerBlock = 8;   // 256 threads = 8 wave32s
    k_gemm_wmma<CPW><<<(totalWaves + wavesPerBlock - 1) / wavesPerBlock, 256, 0, stream>>>(
        X16, WT16, Hbuf, N, K, M);

    int nh = N * GAT_HEADS;
    k_attn<<<(nh + T - 1) / T, T, 0, stream>>>(Hbuf, attS, attD, ASRC, ADST, N, C);

    int nm = N * M;
    k_init<<<(nm + T - 1) / T, T, 0, stream>>>(AMAX, DEN, AGG, nh, nm);

    int te = (E + N) * GAT_HEADS;
    k_edge_max<<<(te + T - 1) / T, T, 0, stream>>>(eidx, E, N, ASRC, ADST, AMAX);
    if (C == 32)
        k_edge_scatter<32><<<(te + T - 1) / T, T, 0, stream>>>(eidx, E, N, ASRC, ADST, AMAX,
                                                               Hbuf, DEN, AGG);
    else
        k_edge_scatter<40><<<(te + T - 1) / T, T, 0, stream>>>(eidx, E, N, ASRC, ADST, AMAX,
                                                               Hbuf, DEN, AGG);
    if (!isFinal)
        k_finalize_concat<<<(nm + T - 1) / T, T, 0, stream>>>(AGG, DEN, bias, Fout, embOut, N, C);
    else
        k_finalize_mean_lsm<<<(N + T - 1) / T, T, 0, stream>>>(AGG, DEN, bias, logitsOut, N, C);
}

extern "C" void kernel_launch(void* const* d_in, const int* in_sizes, int n_in,
                              void* d_out, int out_size, void* d_ws, size_t ws_size,
                              hipStream_t stream) {
    const float* x    = (const float*)d_in[0];
    const int*   eidx = (const int*)d_in[1];   // [2, E] int32 (JAX x64 off)
    const float* W1  = (const float*)d_in[2];
    const float* as1 = (const float*)d_in[3];
    const float* ad1 = (const float*)d_in[4];
    const float* b1  = (const float*)d_in[5];
    const float* W2  = (const float*)d_in[6];
    const float* as2 = (const float*)d_in[7];
    const float* ad2 = (const float*)d_in[8];
    const float* b2  = (const float*)d_in[9];
    const float* W3  = (const float*)d_in[10];
    const float* as3 = (const float*)d_in[11];
    const float* ad3 = (const float*)d_in[12];
    const float* b3  = (const float*)d_in[13];

    const int NFEAT = 128, NHID = 32, NCLASS = 40;
    int N = in_sizes[0] / NFEAT;
    int E = in_sizes[1] / 2;

    // workspace carve-up (256B-aligned slices)
    char* p = (char*)d_ws;
    auto carve = [&](size_t bytes) -> void* {
        void* r = (void*)p;
        p += (bytes + 255) & ~(size_t)255;
        return r;
    };
    _Float16* X16  = (_Float16*)carve((size_t)N * 128 * sizeof(_Float16));
    _Float16* WT16 = (_Float16*)carve((size_t)128 * 160 * sizeof(_Float16));
    float*    Hb   = (float*)carve((size_t)N * 160 * sizeof(float));
    float*    AGG  = (float*)carve((size_t)N * 160 * sizeof(float));
    float*    F    = (float*)carve((size_t)N * 128 * sizeof(float));
    float*    ASRC = (float*)carve((size_t)N * GAT_HEADS * sizeof(float));
    float*    ADST = (float*)carve((size_t)N * GAT_HEADS * sizeof(float));
    unsigned* AMAX = (unsigned*)carve((size_t)N * GAT_HEADS * sizeof(unsigned));
    float*    DEN  = (float*)carve((size_t)N * GAT_HEADS * sizeof(float));

    float* logits = (float*)d_out;                      // [N, NCLASS]
    float* emb    = (float*)d_out + (size_t)N * NCLASS; // [N, 128]

    // Layer 1: 128 -> 4x32 concat, ReLU
    launch_layer(x, W1, as1, ad1, b1, eidx, N, E, NFEAT, NHID, false,
                 X16, WT16, Hb, AGG, ASRC, ADST, AMAX, DEN, F, nullptr, nullptr, stream);
    // Layer 2: 128 -> 4x32 concat, ReLU; output doubles as embeddings
    launch_layer(F, W2, as2, ad2, b2, eidx, N, E, GAT_HEADS * NHID, NHID, false,
                 X16, WT16, Hb, AGG, ASRC, ADST, AMAX, DEN, F, emb, nullptr, stream);
    // Layer 3: 128 -> 4x40 mean, + bias, log_softmax
    launch_layer(F, W3, as3, ad3, b3, eidx, N, E, GAT_HEADS * NHID, NCLASS, true,
                 X16, WT16, Hb, AGG, ASRC, ADST, AMAX, DEN, nullptr, nullptr, logits, stream);
}
